// RotarySelfAttention_22393959482152
// MI455X (gfx1250) — matrix operand order repaired and checked by reading, not compiled, not run on hardware
//
#include <hip/hip_runtime.h>

// ---------------------------------------------------------------------------
// RotarySelfAttention for MI455X (gfx1250): bf16 WMMA everywhere.
// B=2, T=2048, C=2048, H=16, KV=4, D=128.
// Round 2: Kstep=64 GEMMs (half the barriers), async global->LDS copies for
// attention Q/K tiles (ASYNCcnt path), prefetch of next GEMM tiles.
// ---------------------------------------------------------------------------

#define B_   2
#define T_   2048
#define C_   2048
#define NH   16
#define NKV  4
#define HD   128
#define EPS_ 1e-5f
#define SCALE_ 0.08838834764831845f   // 1/sqrt(128)

typedef __attribute__((ext_vector_type(16))) __bf16 v16bf;
typedef __attribute__((ext_vector_type(8)))  float  v8f;

__device__ __forceinline__ __bf16 f2bf(float f) {
  unsigned u = __builtin_bit_cast(unsigned, f);
  u += 0x7FFFu + ((u >> 16) & 1u);                // round-to-nearest-even
  unsigned short h = (unsigned short)(u >> 16);
  return __builtin_bit_cast(__bf16, h);
}

// Async global->LDS 16-byte copy (gfx1250, tracked by ASYNCcnt).
// lds_off: byte offset within LDS (addr[31:0] of the generic shared pointer).
__device__ __forceinline__ void async_b128(unsigned lds_off, const void* g) {
  asm volatile("global_load_async_to_lds_b128 %0, %1, off"
               :: "v"(lds_off), "v"(g) : "memory");
}
__device__ __forceinline__ void wait_async0() {
  asm volatile("s_wait_asynccnt 0x0" ::: "memory");
}

// A-matrix fragment (16x32 bf16, row-major in LDS with `stride` elems/row).
// Lane l: M = l&15, K = (l>>4)*16 + j  (j = 0..15), contiguous in LDS.
__device__ __forceinline__ v16bf fragA(const __bf16* base, int stride, int lane) {
  const __bf16* p = base + (lane & 15) * stride + ((lane >> 4) << 4);
  v16bf a;
#pragma unroll
  for (int j = 0; j < 16; ++j) a[j] = p[j];
  return a;
}

// B-matrix fragment (32x16 bf16) from a K-transposed LDS tile Bt[N][K].
// Lane l: N = l&15, K = (l>>4)*16 + j  (j = 0..15), contiguous in LDS.
__device__ __forceinline__ v16bf fragB(const __bf16* base, int stride, int lane) {
  const __bf16* p = base + (lane & 15) * stride + ((lane >> 4) << 4);
  v16bf b;
#pragma unroll
  for (int j = 0; j < 16; ++j) b[j] = p[j];
  return b;
}

#define WMMA_BF16(a, b, c) \
  __builtin_amdgcn_wmma_f32_16x16x32_bf16(false, (a), false, (b), (short)0, (c), false, false)

// C/D tile mapping: element r of v8f, lane l -> M = r + ((l>>4)<<3), N = l&15.

// ---------------------------------------------------------------------------
// Kernel 1: QKV projection GEMM (fp32 in -> bf16 out), 128x128 block tile,
// Kstep=64, 8 waves, each wave = 2x4 grid of 16x16x32 WMMA tiles. mode==1
// fuses per-head RMSNorm + RoPE in the epilogue (N-tile == one head).
// ---------------------------------------------------------------------------
__global__ __launch_bounds__(256) void qkv_gemm_kernel(
    const float* __restrict__ A, const float* __restrict__ W,
    __bf16* __restrict__ Out, const float* __restrict__ normw,
    const float* __restrict__ fcos, const float* __restrict__ fsin,
    int M, int K, int N, int mode)
{
  extern __shared__ char smem[];
  __bf16* As  = (__bf16*)smem;              // 128 x 72 bf16  (18432 B)
  __bf16* Bt  = (__bf16*)(smem + 18432);    // 128 x 72 bf16  (N-major, K minor)
  float*  Sep = (float*) (smem + 36864);    // 128 x 132 fp32 (epilogue staging)

  const int tid  = threadIdx.x;
  const int lane = tid & 31;
  const int wv   = tid >> 5;
  const int wm   = wv >> 1;      // 0..3 -> rows wm*32
  const int wn   = wv & 1;       // 0..1 -> cols wn*64
  const int m0   = blockIdx.y * 128;
  const int n0   = blockIdx.x * 128;

  v8f acc[2][4] = {};

  for (int k0 = 0; k0 < K; k0 += 64) {
    for (int e = tid; e < 128 * 64; e += 256) {         // A tile fp32 -> bf16
      int r = e >> 6, c = e & 63;
      As[r * 72 + c] = f2bf(A[(size_t)(m0 + r) * K + (k0 + c)]);
    }
    for (int e = tid; e < 64 * 128; e += 256) {         // W tile, transposed
      int kk = e >> 7, n = e & 127;
      Bt[n * 72 + kk] = f2bf(W[(size_t)(k0 + kk) * N + (n0 + n)]);
    }
    if (k0 + 64 < K) {                                  // pull next tiles to L2
      __builtin_prefetch(&A[(size_t)(m0 + (tid >> 1)) * K + k0 + 64], 0, 1);
      __builtin_prefetch(&W[(size_t)(k0 + 64 + (tid & 63)) * N + n0], 0, 1);
    }
    __syncthreads();

#pragma unroll
    for (int dk = 0; dk < 2; ++dk) {
      v16bf af[2], bfr[4];
#pragma unroll
      for (int tm = 0; tm < 2; ++tm)
        af[tm]  = fragA(As + (wm * 32 + tm * 16) * 72 + dk * 32, 72, lane);
#pragma unroll
      for (int tn = 0; tn < 4; ++tn)
        bfr[tn] = fragB(Bt + (wn * 64 + tn * 16) * 72 + dk * 32, 72, lane);
#pragma unroll
      for (int tm = 0; tm < 2; ++tm)
#pragma unroll
        for (int tn = 0; tn < 4; ++tn)
          acc[tm][tn] = WMMA_BF16(af[tm], bfr[tn], acc[tm][tn]);
    }
    __syncthreads();
  }

  if (mode == 0) {                                      // plain bf16 store (V path)
#pragma unroll
    for (int tm = 0; tm < 2; ++tm)
#pragma unroll
      for (int tn = 0; tn < 4; ++tn)
#pragma unroll
        for (int r = 0; r < 8; ++r) {
          int rl = wm * 32 + tm * 16 + ((lane >> 4) << 3) + r;
          int cl = wn * 64 + tn * 16 + (lane & 15);
          Out[(size_t)(m0 + rl) * N + (n0 + cl)] = f2bf(acc[tm][tn][r]);
        }
    return;
  }

  // mode 1: fused RMSNorm + RoPE over this head (N-tile == one head, width 128)
#pragma unroll
  for (int tm = 0; tm < 2; ++tm)
#pragma unroll
    for (int tn = 0; tn < 4; ++tn)
#pragma unroll
      for (int r = 0; r < 8; ++r) {
        int rl = wm * 32 + tm * 16 + ((lane >> 4) << 3) + r;
        int cl = wn * 64 + tn * 16 + (lane & 15);
        Sep[rl * 132 + cl] = acc[tm][tn][r];
      }
  __syncthreads();

  if (tid < 128) {
    const int rl   = tid;
    const int grow = m0 + rl;
    const int t    = grow % T_;
    float ss = 0.f;
    for (int c = 0; c < 128; ++c) { float v = Sep[rl * 132 + c]; ss += v * v; }
    const float rms = rsqrtf(ss * (1.0f / 128.0f) + EPS_);
    const size_t o = (size_t)grow * N + n0;
    for (int j = 0; j < 64; ++j) {
      float xr  = Sep[rl * 132 + 2 * j]     * rms * normw[2 * j];
      float xi  = Sep[rl * 132 + 2 * j + 1] * rms * normw[2 * j + 1];
      float cth = fcos[t * 64 + j];
      float sth = fsin[t * 64 + j];
      Out[o + 2 * j]     = f2bf(xr * cth - xi * sth);
      Out[o + 2 * j + 1] = f2bf(xr * sth + xi * cth);
    }
  }
}

// ---------------------------------------------------------------------------
// Kernel 2: causal flash attention. One block per (b, head, 128-query tile).
// Q/K tiles filled with GLOBAL_LOAD_ASYNC_TO_LDS_B128 (ASYNCcnt), V transposed
// via the regular path. S = Q K^T -> online softmax through LDS -> O += P V.
// ---------------------------------------------------------------------------
__global__ __launch_bounds__(256) void attn_kernel(
    const __bf16* __restrict__ Q, const __bf16* __restrict__ Kk,
    const __bf16* __restrict__ V, __bf16* __restrict__ Y)
{
  extern __shared__ char smem[];
  __bf16* Qs   = (__bf16*)smem;                       // 128 x 136 bf16 (34816 B)
  __bf16* Ks   = (__bf16*)(smem + 34816);             // K tile, row-major [key][d]
  __bf16* Vt   = (__bf16*)(smem + 2 * 34816);         // V tile transposed [d][key]
  __bf16* Ps   = (__bf16*)(smem + 3 * 34816);         // probabilities bf16
  float*  Sf   = (float*) (smem + 4 * 34816);         // 128 x 132 fp32 scores
  float*  rowM = (float*) (smem + 4 * 34816 + 67584); // running max
  float*  rowL = rowM + 128;                          // running denom
  float*  rowA = rowL + 128;                          // correction factor

  const unsigned ldsQ = (unsigned)(uintptr_t)Qs;      // LDS byte offsets (addr[31:0])
  const unsigned ldsK = (unsigned)(uintptr_t)Ks;

  const int tid = threadIdx.x, lane = tid & 31, wv = tid >> 5;
  const int wm = wv >> 1, wn = wv & 1;
  const int qt  = blockIdx.x;          // query tile 0..15
  const int h   = blockIdx.y;          // head 0..15
  const int b   = blockIdx.z;
  const int kvh = h >> 2;              // GQA: 4 query heads share one KV head
  const int q0  = qt * 128;

  // Q tile: 128 rows x 256 B, async copied in 16 B chunks (row stride 272 B).
  for (int e = tid; e < 128 * 16; e += 256) {
    int r = e >> 4, seg = e & 15;
    async_b128(ldsQ + (unsigned)(r * 272 + seg * 16),
               Q + ((size_t)(b * T_ + q0 + r) * NH + h) * HD + seg * 8);
  }
  if (tid < 128) { rowM[tid] = -3.0e38f; rowL[tid] = 0.f; }

  v8f O[2][4] = {};
  wait_async0();
  __syncthreads();

  for (int kt = 0; kt <= qt; ++kt) {
    const int k0 = kt * 128;
    for (int e = tid; e < 128 * 16; e += 256) {        // K tile, async to LDS
      int r = e >> 4, seg = e & 15;
      async_b128(ldsK + (unsigned)(r * 272 + seg * 16),
                 Kk + ((size_t)(b * T_ + k0 + r) * NKV + kvh) * HD + seg * 8);
    }
    for (int e = tid; e < 128 * 128; e += 256) {       // transpose V into LDS
      int r = e >> 7, d = e & 127;
      Vt[d * 136 + r] = V[((size_t)(b * T_ + k0 + r) * NKV + kvh) * HD + d];
    }
    wait_async0();
    __syncthreads();

    // S = Q K^T over D=128 (4 chunks of 32)
    v8f S[2][4] = {};
#pragma unroll
    for (int dk = 0; dk < 4; ++dk) {
      v16bf af[2], bfr[4];
#pragma unroll
      for (int tm = 0; tm < 2; ++tm) af[tm]  = fragA(Qs + (wm * 32 + tm * 16) * 136 + dk * 32, 136, lane);
#pragma unroll
      for (int tn = 0; tn < 4; ++tn) bfr[tn] = fragB(Ks + (wn * 64 + tn * 16) * 136 + dk * 32, 136, lane);
#pragma unroll
      for (int tm = 0; tm < 2; ++tm)
#pragma unroll
        for (int tn = 0; tn < 4; ++tn)
          S[tm][tn] = WMMA_BF16(af[tm], bfr[tn], S[tm][tn]);
    }

    // scale + causal mask -> fp32 LDS
#pragma unroll
    for (int tm = 0; tm < 2; ++tm)
#pragma unroll
      for (int tn = 0; tn < 4; ++tn)
#pragma unroll
        for (int r = 0; r < 8; ++r) {
          int rl = wm * 32 + tm * 16 + ((lane >> 4) << 3) + r;
          int cl = wn * 64 + tn * 16 + (lane & 15);
          float v = S[tm][tn][r] * SCALE_;
          if (k0 + cl > q0 + rl) v = -3.0e38f;
          Sf[rl * 132 + cl] = v;
        }
    __syncthreads();

    // online-softmax row update (one thread per query row)
    if (tid < 128) {
      const int rl = tid;
      float mprev = rowM[rl];
      float mmax = mprev;
      for (int c = 0; c < 128; ++c) mmax = fmaxf(mmax, Sf[rl * 132 + c]);
      float alpha = __expf(mprev - mmax);
      float sum = 0.f;
      for (int c = 0; c < 128; ++c) {
        float p = __expf(Sf[rl * 132 + c] - mmax);
        Ps[rl * 136 + c] = f2bf(p);
        sum += p;
      }
      rowL[rl] = rowL[rl] * alpha + sum;
      rowM[rl] = mmax;
      rowA[rl] = alpha;
    }
    __syncthreads();

    // rescale O by alpha, then O += P @ V  (K-dim = 128 keys, 4 chunks)
#pragma unroll
    for (int tm = 0; tm < 2; ++tm)
#pragma unroll
      for (int r = 0; r < 8; ++r) {
        int rl = wm * 32 + tm * 16 + ((lane >> 4) << 3) + r;
        float a = rowA[rl];
#pragma unroll
        for (int tn = 0; tn < 4; ++tn) O[tm][tn][r] *= a;
      }
#pragma unroll
    for (int kc = 0; kc < 4; ++kc) {
      v16bf af[2], bfr[4];
#pragma unroll
      for (int tm = 0; tm < 2; ++tm) af[tm]  = fragA(Ps + (wm * 32 + tm * 16) * 136 + kc * 32, 136, lane);
#pragma unroll
      for (int tn = 0; tn < 4; ++tn) bfr[tn] = fragB(Vt + (wn * 64 + tn * 16) * 136 + kc * 32, 136, lane);
#pragma unroll
      for (int tm = 0; tm < 2; ++tm)
#pragma unroll
        for (int tn = 0; tn < 4; ++tn)
          O[tm][tn] = WMMA_BF16(af[tm], bfr[tn], O[tm][tn]);
    }
    __syncthreads();
  }

  // finalize: O / l, store bf16 (b,t,h,d)
#pragma unroll
  for (int tm = 0; tm < 2; ++tm)
#pragma unroll
    for (int tn = 0; tn < 4; ++tn)
#pragma unroll
      for (int r = 0; r < 8; ++r) {
        int rl = wm * 32 + tm * 16 + ((lane >> 4) << 3) + r;
        int cl = wn * 64 + tn * 16 + (lane & 15);
        float v = O[tm][tn][r] / rowL[rl];
        Y[((size_t)(b * T_ + q0 + rl) * NH + h) * HD + cl] = f2bf(v);
      }
}

// ---------------------------------------------------------------------------
// Kernel 3: output projection (bf16 A x fp32 W -> fp32 out), Kstep=64.
// ---------------------------------------------------------------------------
__global__ __launch_bounds__(256) void out_gemm_kernel(
    const __bf16* __restrict__ A, const float* __restrict__ W,
    float* __restrict__ Out, int M, int K, int N)
{
  extern __shared__ char smem[];
  __bf16* As = (__bf16*)smem;              // 128 x 72
  __bf16* Bt = (__bf16*)(smem + 18432);    // 128 x 72

  const int tid = threadIdx.x, lane = tid & 31, wv = tid >> 5;
  const int wm = wv >> 1, wn = wv & 1;
  const int m0 = blockIdx.y * 128;
  const int n0 = blockIdx.x * 128;

  v8f acc[2][4] = {};

  for (int k0 = 0; k0 < K; k0 += 64) {
    for (int e = tid; e < 128 * 64; e += 256) {
      int r = e >> 6, c = e & 63;
      As[r * 72 + c] = A[(size_t)(m0 + r) * K + (k0 + c)];
    }
    for (int e = tid; e < 64 * 128; e += 256) {
      int kk = e >> 7, n = e & 127;
      Bt[n * 72 + kk] = f2bf(W[(size_t)(k0 + kk) * N + (n0 + n)]);
    }
    if (k0 + 64 < K) {
      __builtin_prefetch(&A[(size_t)(m0 + (tid >> 1)) * K + k0 + 64], 0, 1);
      __builtin_prefetch(&W[(size_t)(k0 + 64 + (tid & 63)) * N + n0], 0, 1);
    }
    __syncthreads();

#pragma unroll
    for (int dk = 0; dk < 2; ++dk) {
      v16bf af[2], bfr[4];
#pragma unroll
      for (int tm = 0; tm < 2; ++tm)
        af[tm]  = fragA(As + (wm * 32 + tm * 16) * 72 + dk * 32, 72, lane);
#pragma unroll
      for (int tn = 0; tn < 4; ++tn)
        bfr[tn] = fragB(Bt + (wn * 64 + tn * 16) * 72 + dk * 32, 72, lane);
#pragma unroll
      for (int tm = 0; tm < 2; ++tm)
#pragma unroll
        for (int tn = 0; tn < 4; ++tn)
          acc[tm][tn] = WMMA_BF16(af[tm], bfr[tn], acc[tm][tn]);
    }
    __syncthreads();
  }

#pragma unroll
  for (int tm = 0; tm < 2; ++tm)
#pragma unroll
    for (int tn = 0; tn < 4; ++tn)
#pragma unroll
      for (int r = 0; r < 8; ++r) {
        int rl = wm * 32 + tm * 16 + ((lane >> 4) << 3) + r;
        int cl = wn * 64 + tn * 16 + (lane & 15);
        Out[(size_t)(m0 + rl) * N + (n0 + cl)] = acc[tm][tn][r];
      }
}

// ---------------------------------------------------------------------------
extern "C" void kernel_launch(void* const* d_in, const int* in_sizes, int n_in,
                              void* d_out, int out_size, void* d_ws, size_t ws_size,
                              hipStream_t stream)
{
  const float* x    = (const float*)d_in[0];
  const float* fcos = (const float*)d_in[1];   // (T, 64)
  const float* fsin = (const float*)d_in[2];
  const float* wq   = (const float*)d_in[3];   // (C, 2048)
  const float* wk   = (const float*)d_in[4];   // (C, 512)
  const float* wv   = (const float*)d_in[5];   // (C, 512)
  const float* wo   = (const float*)d_in[6];   // (2048, C)
  const float* qnw  = (const float*)d_in[7];
  const float* knw  = (const float*)d_in[8];
  float* out = (float*)d_out;

  const size_t M = (size_t)B_ * T_;            // 4096 rows
  // bf16 workspace: Q(M*2048) K(M*512) V(M*512) Y(M*2048) = 40 MB total
  __bf16* qb = (__bf16*)d_ws;
  __bf16* kb = qb + M * 2048;
  __bf16* vb = kb + M * 512;
  __bf16* yb = vb + M * 512;

  const int smemProj = 36864 + 128 * 132 * 4;              // 104448 B
  const int smemAttn = 4 * 34816 + 67584 + 3 * 128 * 4;    // 208384 B (<320KB WGP LDS)
  const int smemOut  = 36864;

  dim3 blk(256);
  // Q: norm+rope epilogue
  qkv_gemm_kernel<<<dim3(16, 32), blk, smemProj, stream>>>(
      x, wq, qb, qnw, fcos, fsin, 4096, 2048, 2048, 1);
  // K: norm+rope epilogue
  qkv_gemm_kernel<<<dim3(4, 32), blk, smemProj, stream>>>(
      x, wk, kb, knw, fcos, fsin, 4096, 2048, 512, 1);
  // V: plain
  qkv_gemm_kernel<<<dim3(4, 32), blk, smemProj, stream>>>(
      x, wv, vb, nullptr, nullptr, nullptr, 4096, 2048, 512, 0);
  // attention: (query tile, head, batch)
  attn_kernel<<<dim3(16, 16, 2), blk, smemAttn, stream>>>(qb, kb, vb, yb);
  // output projection -> fp32
  out_gemm_kernel<<<dim3(16, 32), blk, smemOut, stream>>>(yb, wo, out, 4096, 2048, 2048);
}
